// PeripheralDWConv2d_2422361555045
// MI455X (gfx1250) — matrix operand order repaired and checked by reading, not compiled, not run on hardware
//
#include <hip/hip_runtime.h>

typedef __attribute__((ext_vector_type(16))) __bf16 v16bf;
typedef __attribute__((ext_vector_type(8)))  __bf16 v8bf;
typedef __attribute__((ext_vector_type(2)))  __bf16 v2bf;
typedef __attribute__((ext_vector_type(8)))  float  v8f;
typedef __attribute__((ext_vector_type(8)))  int    v8i;

#define KSZ 51
#define PADR 25
#define PP 13
#define HW 64
#define DIMC 128
#define NB 16
#define PROWS 114       // 64 + 2*25 padded rows
#define PSTR 152        // bf16 row stride: covers cols 0..143 (x0max+K-1), 16B-aligned rows
#define TW 112          // paired-dword Toeplitz row stride: t in [0,112), j = t-15 in [-15,96]
#define NTHREADS 512

// Recompute the reference's IDX_MAP analytically.
// coords = {0,1,2,4,8,16,25}; signed = [-25,-16,-8,-4,-2,-1,0,1,2,4,8,16,25]
__device__ __forceinline__ int comp_idx(int i) {   // i in [0, 51)
  int o = i - PADR;
  int a = o < 0 ? -o : o;
  int q = (a >= 25) ? 25 : (a >= 16) ? 16 : (a >= 8) ? 8 : (a >= 4) ? 4 : (a >= 2) ? 2 : a;
  int p = (q >= 25) ? 6  : (q >= 16) ? 5  : (q >= 8) ? 4 : (q >= 4) ? 3 : q;
  return (o < 0) ? (6 - p) : (6 + p);
}

__device__ __forceinline__ unsigned pack_pair(float a, float b, float* ra, float* rb) {
  __bf16 ha = (__bf16)a, hb = (__bf16)b;
  *ra = a - (float)ha;
  *rb = b - (float)hb;
  v2bf v = {ha, hb};
  return __builtin_bit_cast(unsigned, v);
}

__global__ __launch_bounds__(NTHREADS)
void pdwconv51_wmma_bf16x(const float* __restrict__ x,
                          const float* __restrict__ wcomp,
                          const float* __restrict__ kpe,
                          float* __restrict__ out) {
  extern __shared__ char smem[];
  __bf16*   plane_hi = (__bf16*)smem;                              // PROWS*PSTR
  __bf16*   plane_lo = plane_hi + PROWS * PSTR;                    // PROWS*PSTR
  unsigned* toep_hi  = (unsigned*)(plane_lo + PROWS * PSTR);       // KSZ*TW dwords
  unsigned* toep_lo  = toep_hi + KSZ * TW;                         // KSZ*TW dwords

  const int pidx = blockIdx.x;                 // flattened (b*DIMC + c)
  const int c    = pidx & (DIMC - 1);
  const int tid  = threadIdx.x;

  // ---- stage zero-padded input plane as split-bf16 (hi + residual lo) -------
  const float* xp = x + (size_t)pidx * HW * HW;
  for (int t = tid; t < PROWS * PSTR; t += NTHREADS) {
    int pr = t / PSTR, pc = t - pr * PSTR;
    int y = pr - PADR, xx = pc - PADR;
    float v = 0.0f;
    if ((unsigned)y < HW && (unsigned)xx < HW) v = xp[y * HW + xx];
    __bf16 h = (__bf16)v;
    plane_hi[t] = h;
    plane_lo[t] = (__bf16)(v - (float)h);
  }
  // ---- build paired-dword Toeplitz rows: d[i][t] = (w[i][t-15], w[i][t-14]) -
  const float* wcc = wcomp + c * (PP * PP);
  for (int t = tid; t < KSZ * TW; t += NTHREADS) {
    int i = t / TW, tt = t - i * TW;
    int j0 = tt - 15;
    int ci = comp_idx(i) * PP;
    float w0 = 0.0f, w1 = 0.0f;
    if ((unsigned)j0 < KSZ)       w0 = wcc[ci + comp_idx(j0)]     + kpe[i * KSZ + j0];
    if ((unsigned)(j0 + 1) < KSZ) w1 = wcc[ci + comp_idx(j0 + 1)] + kpe[i * KSZ + j0 + 1];
    float r0, r1;
    toep_hi[t] = pack_pair(w0, w1, &r0, &r1);
    v2bf lo = {(__bf16)r0, (__bf16)r1};
    toep_lo[t] = __builtin_bit_cast(unsigned, lo);
  }
  __syncthreads();

  // ---- WMMA main loop: out = Ah*Bh + Ah*Bl + Al*Bh, K = 96 (3 steps of 32) --
  const int lane  = tid & 31;
  const int wave  = tid >> 5;                  // 16 waves -> 4x4 tiles of 16x16
  const int ty = wave >> 2, tx = wave & 3;
  const int m     = lane & 15;                 // A: M row / B: N col
  const int koffA = (lane >> 4) << 4;          // A K half {0,16}
  const int koffB = (lane >> 4) << 4;          // B K half {0,16}

  v8f ahh = {}, ahl = {}, alh = {};
  const __bf16*   aH = plane_hi + (ty * 16 + m) * PSTR + tx * 16 + koffA;
  const __bf16*   aL = plane_lo + (ty * 16 + m) * PSTR + tx * 16 + koffA;
  const unsigned* bH = toep_hi + 15 + koffB - m;
  const unsigned* bL = toep_lo + 15 + koffB - m;

  for (int i = 0; i < KSZ; ++i) {
    const __bf16*   arH = aH + i * PSTR;
    const __bf16*   arL = aL + i * PSTR;
    const unsigned* brH = bH + i * TW;
    const unsigned* brL = bL + i * TW;
#pragma unroll
    for (int ks = 0; ks < 3; ++ks) {
      const int kb = ks * 32;
      // A fragments: two aligned b128 LDS loads per operand
      v8bf h0 = *(const v8bf*)(arH + kb);
      v8bf h1 = *(const v8bf*)(arH + kb + 8);
      v8bf l0 = *(const v8bf*)(arL + kb);
      v8bf l1 = *(const v8bf*)(arL + kb + 8);
      v16bf afH = __builtin_shufflevector(h0, h1, 0,1,2,3,4,5,6,7,8,9,10,11,12,13,14,15);
      v16bf afL = __builtin_shufflevector(l0, l1, 0,1,2,3,4,5,6,7,8,9,10,11,12,13,14,15);
      // B fragments: 8 paired dwords each (VGPR r = K pair (kb+koffB+2r, +1) for col m)
      v8i biH, biL;
#pragma unroll
      for (int r = 0; r < 8; ++r) {
        biH[r] = (int)brH[kb + 2 * r];
        biL[r] = (int)brL[kb + 2 * r];
      }
      v16bf bfH = __builtin_bit_cast(v16bf, biH);
      v16bf bfL = __builtin_bit_cast(v16bf, biL);

      ahh = __builtin_amdgcn_wmma_f32_16x16x32_bf16(false, afH, false, bfH, (short)0, ahh, false, false);
      ahl = __builtin_amdgcn_wmma_f32_16x16x32_bf16(false, afH, false, bfL, (short)0, ahl, false, false);
      alh = __builtin_amdgcn_wmma_f32_16x16x32_bf16(false, afL, false, bfH, (short)0, alh, false, false);
    }
  }
  v8f acc = ahh + ahl + alh;

  // ---- store C: VGPR r -> M = r + 8*(lane>=16), N = lane%16 -----------------
  float* op = out + (size_t)pidx * HW * HW;
  const int rbase = ty * 16 + ((lane >> 4) << 3);
  const int col   = tx * 16 + m;
#pragma unroll
  for (int r = 0; r < 8; ++r)
    op[(rbase + r) * HW + col] = acc[r];
}

extern "C" void kernel_launch(void* const* d_in, const int* in_sizes, int n_in,
                              void* d_out, int out_size, void* d_ws, size_t ws_size,
                              hipStream_t stream) {
  const float* x     = (const float*)d_in[0];
  const float* wcomp = (const float*)d_in[1];
  const float* kpe   = (const float*)d_in[2];
  float* out = (float*)d_out;

  const int nplanes = NB * DIMC;               // 2048 workgroups
  const size_t smem = (size_t)(2 * PROWS * PSTR * sizeof(__bf16) +
                               2 * KSZ * TW * sizeof(unsigned)); // 115,008 B

  (void)hipFuncSetAttribute((const void*)pdwconv51_wmma_bf16x,
                            hipFuncAttributeMaxDynamicSharedMemorySize, (int)smem);
  hipLaunchKernelGGL(pdwconv51_wmma_bf16x, dim3(nplanes), dim3(NTHREADS), smem, stream,
                     x, wcomp, kpe, out);
}
